// RWKV_Tmix_x070_27908697490173
// MI455X (gfx1250) — compile-verified, matrix operand order checked by reading, not run on hardware
//
#include <hip/hip_runtime.h>
#include <hip/hip_bf16.h>
#include <math.h>

// Problem constants (fixed by the reference)
#define Bd    2
#define Td    4096
#define Cd    2048
#define HSd   64
#define Hd    32
#define NKd   16
#define HKd   128
#define LORAd 8
#define GN_EPSd 0.00064f

typedef __attribute__((ext_vector_type(16))) __bf16 v16bf;
typedef __attribute__((ext_vector_type(8)))  float  v8f;
typedef int v4i_ __attribute__((vector_size(16)));   // matches builtin param pointee

#if __has_builtin(__builtin_amdgcn_global_load_async_to_lds_b128)
#define USE_ASYNC_LDS 1
#else
#define USE_ASYNC_LDS 0
#endif

__device__ __forceinline__ float sigm_(float x) { return 1.0f / (1.0f + expf(-x)); }

__device__ __forceinline__ void wait_async_() {
#if USE_ASYNC_LDS
#if __has_builtin(__builtin_amdgcn_s_wait_asynccnt)
    __builtin_amdgcn_s_wait_asynccnt(0);
#else
    asm volatile("s_wait_asynccnt 0" ::: "memory");
#endif
#endif
}

// Two 16B-aligned LDS halves -> one 32-element-equivalent bf16 fragment
__device__ __forceinline__ v16bf ld_lds_v16bf(const __bf16* p) {
    union { v16bf v; float4 q[2]; } u;
    u.q[0] = *(const float4*)(p);
    u.q[1] = *(const float4*)(p + 8);
    return u.v;
}

// ---------------------------------------------------------------------------
// K0: token shift + mixes + all four LoRA paths.
// One block per token row (b*T+t). 256 threads x 8 channels.
// ---------------------------------------------------------------------------
__global__ __launch_bounds__(256) void k_prep(
    const float* __restrict__ x,
    const float* __restrict__ mx_r, const float* __restrict__ mx_w,
    const float* __restrict__ mx_k, const float* __restrict__ mx_v,
    const float* __restrict__ mx_a, const float* __restrict__ mx_g,
    const float* __restrict__ w0, const float* __restrict__ w1, const float* __restrict__ w2,
    const float* __restrict__ a0, const float* __restrict__ a1, const float* __restrict__ a2,
    const float* __restrict__ v0, const float* __restrict__ v1, const float* __restrict__ v2,
    const float* __restrict__ g1, const float* __restrict__ g2,
    __bf16* __restrict__ xr_bf, __bf16* __restrict__ xk_bf, __bf16* __restrict__ xv_bf,
    float* __restrict__ decay, float* __restrict__ a_sig,
    float* __restrict__ g_out, float* __restrict__ vmix)
{
    const int row = blockIdx.x;      // b*T + t
    const int t   = row % Td;
    const int tid = threadIdx.x;
    const int c0  = tid * 8;
    const size_t base = (size_t)row * Cd;

    __shared__ float red[256 * LORAd];
    __shared__ float hbuf[4][LORAd];

    const float* xrow = x + base;
    float xc[8], xx[8];
#pragma unroll
    for (int i = 0; i < 8; i++) {
        xc[i] = xrow[c0 + i];
        float xp = (t > 0) ? xrow[c0 + i - Cd] : 0.0f;
        xx[i] = xp - xc[i];
    }
    float mr[8], mw[8], mk[8], mv[8], ma[8], mg[8];
#pragma unroll
    for (int i = 0; i < 8; i++) {
        int c = c0 + i;
        mr[i] = xc[i] + xx[i] * mx_r[c];
        mw[i] = xc[i] + xx[i] * mx_w[c];
        mk[i] = xc[i] + xx[i] * mx_k[c];
        mv[i] = xc[i] + xx[i] * mx_v[c];
        ma[i] = xc[i] + xx[i] * mx_a[c];
        mg[i] = xc[i] + xx[i] * mx_g[c];
    }
#pragma unroll
    for (int i = 0; i < 8; i++) {
        size_t idx = base + c0 + i;
        xr_bf[idx] = (__bf16)mr[i];
        xk_bf[idx] = (__bf16)mk[i];
        xv_bf[idx] = (__bf16)mv[i];
    }

    // LoRA stage 1: h_q[l] = sum_c mix_q[c] * M1_q[c,l]
    float accw[LORAd] = {0}, acca[LORAd] = {0}, accv[LORAd] = {0}, accg[LORAd] = {0};
#pragma unroll
    for (int i = 0; i < 8; i++) {
        int c = c0 + i;
#pragma unroll
        for (int l = 0; l < LORAd; l++) {
            accw[l] += mw[i] * w1[(size_t)c * LORAd + l];
            acca[l] += ma[i] * a1[(size_t)c * LORAd + l];
            accv[l] += mv[i] * v1[(size_t)c * LORAd + l];
            accg[l] += mg[i] * g1[(size_t)c * LORAd + l];
        }
    }
    float* accs[4] = {accw, acca, accv, accg};
#pragma unroll
    for (int q = 0; q < 4; q++) {
#pragma unroll
        for (int l = 0; l < LORAd; l++) red[tid * LORAd + l] = accs[q][l];
        __syncthreads();
        for (int s = 128; s > 0; s >>= 1) {
            if (tid < s) {
#pragma unroll
                for (int l = 0; l < LORAd; l++)
                    red[tid * LORAd + l] += red[(tid + s) * LORAd + l];
            }
            __syncthreads();
        }
        if (tid < LORAd) hbuf[q][tid] = red[tid];
        __syncthreads();
    }

    float tw[LORAd], ha[LORAd], hv[LORAd], sg[LORAd];
#pragma unroll
    for (int l = 0; l < LORAd; l++) {
        tw[l] = tanhf(hbuf[0][l]);
        ha[l] = hbuf[1][l];
        hv[l] = hbuf[2][l];
        sg[l] = sigm_(hbuf[3][l]);
    }
#pragma unroll
    for (int i = 0; i < 8; i++) {
        int c = c0 + i;
        size_t idx = base + c;
        float ws = w0[c], av = a0[c], vm = v0[c], gv = 0.0f;
#pragma unroll
        for (int l = 0; l < LORAd; l++) {
            ws += tw[l] * w2[(size_t)l * Cd + c];
            av += ha[l] * a2[(size_t)l * Cd + c];
            vm += hv[l] * v2[(size_t)l * Cd + c];
            gv += sg[l] * g2[(size_t)l * Cd + c];
        }
        float wfin = -log1pf(expf(-ws)) - 0.5f;   // -softplus(-ws) - 0.5
        decay[idx] = expf(-expf(wfin));
        a_sig[idx] = sigm_(av);
        vmix[idx]  = sigm_(vm);
        g_out[idx] = gv;
    }
}

// ---------------------------------------------------------------------------
// f32 -> bf16 cast (weights)
// ---------------------------------------------------------------------------
__global__ void k_cast_bf16(const float* __restrict__ in, __bf16* __restrict__ out, int n)
{
    int i = blockIdx.x * blockDim.x + threadIdx.x;
    if (i < n) out[i] = (__bf16)in[i];
}

// ---------------------------------------------------------------------------
// NT GEMM via v_wmma_f32_16x16x32_bf16 with async-to-LDS staging of A.
// C[n,m] = sum_k A[n,k] * Bw[m,k];  A:[N,K] row-major, Bw:[M,K] row-major.
// Block: 256 thr = 8 waves, tile 64 rows x 256 cols.
// Wave: 32 rows x 64 cols = 2 A-frags x 4 B-frags = 8 WMMA per K-step(32).
// A tile double-buffered in LDS: 64 rows x 32 k, row pitch 40 elems (80B).
// ---------------------------------------------------------------------------
#define GT_ROWS 64
#define GT_COLS 256
#define KSTEP   32
#define APAD    40

__global__ __launch_bounds__(256) void k_gemm_nt(
    const __bf16* __restrict__ A, const __bf16* __restrict__ Bw,
    float* __restrict__ Cout, int N, int K, int M)
{
    __shared__ __bf16 Asm[2 * GT_ROWS * APAD];

    const int tid  = threadIdx.x;
    const int lane = tid & 31;
    const int wave = tid >> 5;
    const int wrow = wave >> 2;                 // 0..1 : 32-row group
    const int wcol = wave & 3;                  // 0..3 : 64-col group
    const int rowBlock = blockIdx.x * GT_ROWS;
    const int colBase  = blockIdx.y * GT_COLS + wcol * 64;
    const int lrow  = lane & 15;
    const int khalf = (lane >> 4) * 16;

    // staging map: thread copies one 16B chunk: row = tid>>2, seg = (tid&3)*8 elems
    const int sr = tid >> 2;
    const int ss = (tid & 3) * 8;
    const __bf16* agsrc = A + (size_t)(rowBlock + sr) * K + ss;
    __bf16* const ldst0 = &Asm[sr * APAD + ss];

    const __bf16* bp = Bw + (size_t)(colBase + lrow) * K + khalf;

    v8f zero = {0.f,0.f,0.f,0.f,0.f,0.f,0.f,0.f};
    v8f acc[2][4];
#pragma unroll
    for (int f = 0; f < 2; f++)
#pragma unroll
        for (int c = 0; c < 4; c++) acc[f][c] = zero;

    auto stage = [&](int kk, int buf) {
        __bf16* dst = ldst0 + buf * (GT_ROWS * APAD);
#if USE_ASYNC_LDS
        __builtin_amdgcn_global_load_async_to_lds_b128(
            (__attribute__((address_space(1))) v4i_*)(agsrc + kk),
            (__attribute__((address_space(3))) v4i_*)dst, 0, 0);
#else
        *(float4*)dst = *(const float4*)(agsrc + kk);
#endif
    };

    const int arow0 = (wrow * 32 + lrow) * APAD + khalf;
    const int arow1 = arow0 + 16 * APAD;

    stage(0, 0);
    wait_async_();
    __syncthreads();

    for (int kk = 0; kk < K; kk += KSTEP) {
        const int cur = (kk >> 5) & 1;
        if (kk + KSTEP < K) stage(kk + KSTEP, cur ^ 1);

        const __bf16* ab = Asm + cur * (GT_ROWS * APAD);
        v16bf a0 = ld_lds_v16bf(ab + arow0);
        v16bf a1 = ld_lds_v16bf(ab + arow1);
#pragma unroll
        for (int c = 0; c < 4; c++) {
            v16bf bfm = *reinterpret_cast<const v16bf*>(bp + (size_t)c * 16 * K + kk);
            acc[0][c] = __builtin_amdgcn_wmma_f32_16x16x32_bf16(
                false, a0, false, bfm, (short)0, acc[0][c], false, false);
            acc[1][c] = __builtin_amdgcn_wmma_f32_16x16x32_bf16(
                false, a1, false, bfm, (short)0, acc[1][c], false, false);
        }

        wait_async_();
        __syncthreads();
    }

    // f32 C/D layout: VGPR j -> row j + 8*(lane>>4), col = lane&15
    const int rr = (lane >> 4) * 8;
    const int cc = colBase + lrow;
#pragma unroll
    for (int f = 0; f < 2; f++) {
        const int rbase = rowBlock + wrow * 32 + f * 16 + rr;
#pragma unroll
        for (int c = 0; c < 4; c++) {
#pragma unroll
            for (int j = 0; j < 8; j++) {
                Cout[(size_t)(rbase + j) * M + cc + c * 16] = acc[f][c][j];
            }
        }
    }
}

// ---------------------------------------------------------------------------
// K2.5: v blend, kk normalize (per 64-group), build a=-kk, b=kk*a, final k.
// ---------------------------------------------------------------------------
__global__ __launch_bounds__(256) void k_post(
    float* __restrict__ k, float* __restrict__ v,
    const float* __restrict__ v_first, const float* __restrict__ vmix,
    const float* __restrict__ a_sig,
    const float* __restrict__ k_k, const float* __restrict__ k_a,
    float* __restrict__ aa, float* __restrict__ bb)
{
    const int row = blockIdx.x;
    const int tid = threadIdx.x;
    const int c0  = tid * 8;
    const size_t base = (size_t)row * Cd;
    __shared__ float gred[256];

    float kv[8], av[8], kkv[8];
    float ss = 0.f;
#pragma unroll
    for (int i = 0; i < 8; i++) {
        int c = c0 + i; size_t idx = base + c;
        kv[i]  = k[idx];
        av[i]  = a_sig[idx];
        kkv[i] = kv[i] * k_k[c];
        ss += kkv[i] * kkv[i];
    }
    gred[tid] = ss;
    __syncthreads();
    float gs = 0.f;
    const int gb = tid & ~7;            // 8 threads == one 64-channel group
#pragma unroll
    for (int u = 0; u < 8; u++) gs += gred[gb + u];
    const float inv = 1.0f / fmaxf(sqrtf(gs), 1e-8f);

#pragma unroll
    for (int i = 0; i < 8; i++) {
        int c = c0 + i; size_t idx = base + c;
        float kkn = kkv[i] * inv;
        aa[idx] = -kkn;
        bb[idx] = kkn * av[i];
        k[idx]  = kv[i] * (1.0f + (av[i] - 1.0f) * k_a[c]);
        float vv = v[idx], vf = v_first[idx], vm = vmix[idx];
        v[idx] = vv + (vf - vv) * vm;
    }
}

// ---------------------------------------------------------------------------
// K3: RWKV-7 scan. One wave32 per (b, head). S is 16x16 f32 = 8 regs/lane.
// ---------------------------------------------------------------------------
__global__ __launch_bounds__(32) void k_scan(
    const float* __restrict__ r, const float* __restrict__ decay,
    const float* __restrict__ k, const float* __restrict__ v,
    const float* __restrict__ aa, const float* __restrict__ bb,
    float* __restrict__ o)
{
    const int bh = blockIdx.x;
    const int b = bh / HKd;
    const int h = bh % HKd;
    const int lane  = threadIdx.x;
    const int i     = lane & 15;
    const int jbase = (lane >> 4) * 8;

    float S[8];
#pragma unroll
    for (int j = 0; j < 8; j++) S[j] = 0.f;

    size_t off = (size_t)b * Td * Cd + (size_t)h * NKd + i;
    for (int t = 0; t < Td; t++, off += Cd) {
        float rl = r[off], dl = decay[off], kl = k[off];
        float vl = v[off], al = aa[off],   bl = bb[off];

        float sa = 0.f;
#pragma unroll
        for (int j = 0; j < 8; j++) sa += S[j] * __shfl(al, jbase + j, 32);
        sa += __shfl_xor(sa, 16, 32);

        float y = 0.f;
#pragma unroll
        for (int j = 0; j < 8; j++) {
            float Sn = S[j] * __shfl(dl, jbase + j, 32)
                     + sa   * __shfl(bl, jbase + j, 32)
                     + vl   * __shfl(kl, jbase + j, 32);
            S[j] = Sn;
            y += Sn * __shfl(rl, jbase + j, 32);
        }
        y += __shfl_xor(y, 16, 32);
        if (lane < 16) o[off] = y;
    }
}

// ---------------------------------------------------------------------------
// K4: GroupNorm + rank-1 residual + gating, emit bf16 for output GEMM.
// ---------------------------------------------------------------------------
__global__ __launch_bounds__(256) void k_gnout(
    const float* __restrict__ o, const float* __restrict__ r,
    const float* __restrict__ k, const float* __restrict__ v,
    const float* __restrict__ g, const float* __restrict__ r_k,
    const float* __restrict__ ln_g, const float* __restrict__ ln_b,
    __bf16* __restrict__ og_bf)
{
    const int row = blockIdx.x;
    const int tid = threadIdx.x;
    const int c0  = tid * 8;
    const size_t base = (size_t)row * Cd;
    __shared__ float rA[256], rB[256];

    float ov[8], rv[8], kv[8], vv[8];
    float s1 = 0.f, s2 = 0.f, sd = 0.f;
#pragma unroll
    for (int i = 0; i < 8; i++) {
        int c = c0 + i; size_t idx = base + c;
        ov[i] = o[idx]; rv[i] = r[idx]; kv[i] = k[idx]; vv[i] = v[idx];
        s1 += ov[i];
        s2 += ov[i] * ov[i];
        sd += rv[i] * kv[i] * r_k[c];
    }
    rA[tid] = s1; rB[tid] = s2;
    __syncthreads();
    float gs1 = 0.f, gs2 = 0.f;
    const int gb = tid & ~7;
#pragma unroll
    for (int u = 0; u < 8; u++) { gs1 += rA[gb + u]; gs2 += rB[gb + u]; }
    __syncthreads();
    rA[tid] = sd;
    __syncthreads();
    float gd = 0.f;
#pragma unroll
    for (int u = 0; u < 8; u++) gd += rA[gb + u];

    const float mu   = gs1 * (1.0f / HSd);
    const float var  = gs2 * (1.0f / HSd) - mu * mu;
    const float rstd = rsqrtf(var + GN_EPSd);

#pragma unroll
    for (int i = 0; i < 8; i++) {
        int c = c0 + i; size_t idx = base + c;
        float on = (ov[i] - mu) * rstd * ln_g[c] + ln_b[c];
        on += gd * vv[i];
        og_bf[idx] = (__bf16)(on * g[idx]);
    }
}

// ---------------------------------------------------------------------------
extern "C" void kernel_launch(void* const* d_in, const int* in_sizes, int n_in,
                              void* d_out, int out_size, void* d_ws, size_t ws_size,
                              hipStream_t stream)
{
    (void)in_sizes; (void)n_in; (void)out_size; (void)ws_size;

    const float* x      = (const float*)d_in[0];
    const float* vfirst = (const float*)d_in[1];
    const float* mx_r   = (const float*)d_in[2];
    const float* mx_w   = (const float*)d_in[3];
    const float* mx_k   = (const float*)d_in[4];
    const float* mx_v   = (const float*)d_in[5];
    const float* mx_a   = (const float*)d_in[6];
    const float* mx_g   = (const float*)d_in[7];
    const float* w0     = (const float*)d_in[8];
    const float* w1     = (const float*)d_in[9];
    const float* w2     = (const float*)d_in[10];
    const float* a0     = (const float*)d_in[11];
    const float* a1     = (const float*)d_in[12];
    const float* a2     = (const float*)d_in[13];
    const float* v0     = (const float*)d_in[14];
    const float* v1     = (const float*)d_in[15];
    const float* v2     = (const float*)d_in[16];
    const float* g1     = (const float*)d_in[17];
    const float* g2     = (const float*)d_in[18];
    const float* k_k    = (const float*)d_in[19];
    const float* k_a    = (const float*)d_in[20];
    const float* r_k    = (const float*)d_in[21];
    const float* Wr     = (const float*)d_in[22];
    const float* Wk     = (const float*)d_in[23];
    const float* Wv     = (const float*)d_in[24];
    const float* Wo     = (const float*)d_in[25];
    const float* ln_g   = (const float*)d_in[26];
    const float* ln_b   = (const float*)d_in[27];

    float* out = (float*)d_out;

    const size_t NT  = (size_t)Bd * Td;       // 8192
    const size_t NTC = NT * Cd;
    const size_t CC  = (size_t)Cd * Cd;

    char* ws = (char*)d_ws;
    size_t off = 0;
    auto carve = [&](size_t bytes) -> char* {
        char* p = ws + off;
        off += (bytes + 255) & ~(size_t)255;
        return p;
    };

    __bf16* xr_bf = (__bf16*)carve(NTC * 2);
    __bf16* xk_bf = (__bf16*)carve(NTC * 2);
    __bf16* xv_bf = (__bf16*)carve(NTC * 2);
    __bf16* og_bf = (__bf16*)carve(NTC * 2);
    __bf16* wr_bf = (__bf16*)carve(CC * 2);
    __bf16* wk_bf = (__bf16*)carve(CC * 2);
    __bf16* wv_bf = (__bf16*)carve(CC * 2);
    __bf16* wo_bf = (__bf16*)carve(CC * 2);
    float* rbuf   = (float*)carve(NTC * 4);
    float* kbuf   = (float*)carve(NTC * 4);
    float* vbuf   = (float*)carve(NTC * 4);
    float* dbuf   = (float*)carve(NTC * 4);   // decay
    float* abuf   = (float*)carve(NTC * 4);   // sigmoid a
    float* gbuf   = (float*)carve(NTC * 4);   // gate g
    float* vmixb  = (float*)carve(NTC * 4);
    float* aabuf  = (float*)carve(NTC * 4);   // -kk
    float* bbbuf  = (float*)carve(NTC * 4);   // kk*a
    float* obuf   = (float*)carve(NTC * 4);   // scan output

    // Weight f32 -> bf16
    {
        int n = (int)CC, thr = 256, blk = (n + thr - 1) / thr;
        k_cast_bf16<<<blk, thr, 0, stream>>>(Wr, wr_bf, n);
        k_cast_bf16<<<blk, thr, 0, stream>>>(Wk, wk_bf, n);
        k_cast_bf16<<<blk, thr, 0, stream>>>(Wv, wv_bf, n);
        k_cast_bf16<<<blk, thr, 0, stream>>>(Wo, wo_bf, n);
    }

    // Token shift, mixes, LoRA paths
    k_prep<<<(int)NT, 256, 0, stream>>>(
        x, mx_r, mx_w, mx_k, mx_v, mx_a, mx_g,
        w0, w1, w2, a0, a1, a2, v0, v1, v2, g1, g2,
        xr_bf, xk_bf, xv_bf, dbuf, abuf, gbuf, vmixb);

    // r, k, v projections (WMMA)
    dim3 ggrid((unsigned)(NT / GT_ROWS), (unsigned)(Cd / GT_COLS));
    k_gemm_nt<<<ggrid, 256, 0, stream>>>(xr_bf, wr_bf, rbuf, (int)NT, Cd, Cd);
    k_gemm_nt<<<ggrid, 256, 0, stream>>>(xk_bf, wk_bf, kbuf, (int)NT, Cd, Cd);
    k_gemm_nt<<<ggrid, 256, 0, stream>>>(xv_bf, wv_bf, vbuf, (int)NT, Cd, Cd);

    // v blend + kk normalize + a/b vectors + final k
    k_post<<<(int)NT, 256, 0, stream>>>(kbuf, vbuf, vfirst, vmixb, abuf, k_k, k_a,
                                        aabuf, bbbuf);

    // Sequential scan: one wave per (b, head)
    k_scan<<<Bd * HKd, 32, 0, stream>>>(rbuf, dbuf, kbuf, vbuf, aabuf, bbbuf, obuf);

    // GroupNorm + residual + gate -> bf16
    k_gnout<<<(int)NT, 256, 0, stream>>>(obuf, rbuf, kbuf, vbuf, gbuf, r_k, ln_g, ln_b,
                                         og_bf);

    // Output projection (WMMA) -> first half of d_out
    k_gemm_nt<<<ggrid, 256, 0, stream>>>(og_bf, wo_bf, out, (int)NT, Cd, Cd);

    // Second tuple element: pass-through v_first
    (void)hipMemcpyAsync(out + NTC, vfirst, NTC * sizeof(float),
                         hipMemcpyDeviceToDevice, stream);
}
